// MultiheadSelfAttention_65712999628915
// MI455X (gfx1250) — matrix operand order repaired and checked by reading, not compiled, not run on hardware
//
#include <hip/hip_runtime.h>

// MI455X / gfx1250, wave32. All matmuls via V_WMMA_F32_16X16X32_F16.

typedef __attribute__((ext_vector_type(16))) _Float16 v16h;
typedef __attribute__((ext_vector_type(8)))  _Float16 v8h;
typedef __attribute__((ext_vector_type(8)))  float    v8f;

#define T_SEQ 2048
#define BATCH 4
#define EMB   1024
#define HEADS 16
#define HDIM  64
#define MROWS (T_SEQ * BATCH)   // 8192
#define NEGBIG (-1.0e30f)

static __device__ __forceinline__ v8f wmma16(v16h a, v16h b, v8f c) {
  // 8 args: (neg_a, A, neg_b, B, c_mod, C, reuse_a, reuse_b)
  return __builtin_amdgcn_wmma_f32_16x16x32_f16(false, a, false, b, (short)0, c,
                                                false, false);
}

// ---------------------------------------------------------------- converts
__global__ void cvt_f32_to_f16(const float* __restrict__ src,
                               _Float16* __restrict__ dst, int n) {
  int i = blockIdx.x * blockDim.x + threadIdx.x;
  if (i < n) dst[i] = (_Float16)src[i];
}

// ---------------------------------------------------------------- tiled GEMM core
// C[128x128] block, 8 waves (2x4), each wave 64x32 (4x2 tiles of 16x16).
// A: row-major [M][K] f16.  B: row-major [N][K] f16 (i.e. C = A * B^T).
static __device__ __forceinline__ void gemm_block(
    const _Float16* __restrict__ Ag, const _Float16* __restrict__ Bg, int K,
    int m0, int n0, _Float16* lA, _Float16* lB, v8f acc[4][2]) {
  const int tid  = threadIdx.x;
  const int w    = tid >> 5;
  const int lane = tid & 31;
  const int hi   = lane >> 4;
  const int ln   = lane & 15;
  const int wm   = (w >> 2) * 64;
  const int wn   = (w & 3) * 32;
  const v8f vz = {};
#pragma unroll
  for (int mt = 0; mt < 4; ++mt)
#pragma unroll
    for (int nt = 0; nt < 2; ++nt) acc[mt][nt] = vz;

  const int row = tid >> 1;
  const int seg = (tid & 1) * 16;
  for (int k0 = 0; k0 < K; k0 += 32) {
    __syncthreads();
    *(v16h*)(lA + row * 32 + seg) =
        *(const v16h*)(Ag + (size_t)(m0 + row) * K + k0 + seg);
    *(v16h*)(lB + row * 32 + seg) =
        *(const v16h*)(Bg + (size_t)(n0 + row) * K + k0 + seg);
    __syncthreads();
    v16h bf[2];
#pragma unroll
    for (int nt = 0; nt < 2; ++nt)
      bf[nt] = *(const v16h*)(lB + (wn + nt * 16 + ln) * 32 + 16 * hi);
#pragma unroll
    for (int mt = 0; mt < 4; ++mt) {
      union { v16h v; v8h p[2]; } a;
      const _Float16* ap = lA + (wm + mt * 16 + ln) * 32;
      a.p[0] = *(const v8h*)(ap + 16 * hi);
      a.p[1] = *(const v8h*)(ap + 16 * hi + 8);
#pragma unroll
      for (int nt = 0; nt < 2; ++nt)
        acc[mt][nt] = wmma16(a.v, bf[nt], acc[mt][nt]);
    }
  }
}

// ---------------------------------------------------------------- QKV projection
// P[m][f] = X[m][:] . Wi[f][:] + b_in[f]; scatter into Q/K (B,H,T,D) and Vt (B,H,D,T).
__global__ __launch_bounds__(256) void proj_kernel(
    const _Float16* __restrict__ Xh, const _Float16* __restrict__ Wih,
    const float* __restrict__ b_in, _Float16* __restrict__ Qh,
    _Float16* __restrict__ Kh, _Float16* __restrict__ Vth) {
  __shared__ __align__(64) _Float16 lA[128 * 32];
  __shared__ __align__(64) _Float16 lB[128 * 32];
  const int m0 = blockIdx.x * 128;
  const int n0 = blockIdx.y * 128;
  v8f acc[4][2];
  gemm_block(Xh, Wih, EMB, m0, n0, lA, lB, acc);

  const int tid = threadIdx.x, w = tid >> 5, lane = tid & 31;
  const int hi = lane >> 4, ln = lane & 15;
  const int wm = (w >> 2) * 64, wn = (w & 3) * 32;
#pragma unroll
  for (int mt = 0; mt < 4; ++mt)
#pragma unroll
    for (int nt = 0; nt < 2; ++nt)
#pragma unroll
      for (int r = 0; r < 8; ++r) {
        const int m = m0 + wm + mt * 16 + r + 8 * hi;
        const int f = n0 + wn + nt * 16 + ln;
        const float val = acc[mt][nt][r] + b_in[f];
        const int t = m >> 2, b = m & 3;              // m = t*BATCH + b
        const int qkv = f >> 10, e = f & 1023;
        const int h = e >> 6, d = e & 63;
        const int bh = (b << 4) + h;
        if (qkv == 0)
          Qh[((size_t)bh * T_SEQ + t) * HDIM + d] = (_Float16)(val * 0.125f);
        else if (qkv == 1)
          Kh[((size_t)bh * T_SEQ + t) * HDIM + d] = (_Float16)val;
        else
          Vth[((size_t)bh * HDIM + d) * T_SEQ + t] = (_Float16)val;
      }
}

// ---------------------------------------------------------------- fused attention
// One WG per (b*h, 128 query rows). 8 waves x 16 rows. Flash-style online softmax.
__global__ __launch_bounds__(256) void attn_kernel(
    const _Float16* __restrict__ Qh, const _Float16* __restrict__ Kh,
    const _Float16* __restrict__ Vth, const unsigned char* __restrict__ mask,
    _Float16* __restrict__ Attnh) {
  __shared__ __align__(64) _Float16 Ktile[64 * 64];   // [s][d]
  __shared__ __align__(64) _Float16 Vtile[64 * 64];   // [d][s]
  __shared__ __align__(64) _Float16 Pbuf[8 * 16 * 64];// per-wave [16][64]

  const int bh   = blockIdx.x;          // 0..63  (b*16 + h)
  const int tblk = blockIdx.y;          // 0..15
  const int b = bh >> 4, h = bh & 15;
  const int tid = threadIdx.x, w = tid >> 5, lane = tid & 31;
  const int hi = lane >> 4, ln = lane & 15;

  const _Float16* Qbase = Qh  + (size_t)bh * T_SEQ * HDIM;
  const _Float16* Kbase = Kh  + (size_t)bh * T_SEQ * HDIM;
  const _Float16* Vbase = Vth + (size_t)bh * HDIM * T_SEQ;
  const unsigned char* mrow = mask + (size_t)b * T_SEQ;
  const int trow = tblk * 128 + w * 16;

  // Resident Q fragments (16 rows x 64 k): A-fragment layout from row-major.
  v16h qf[2];
  {
    const _Float16* qp = Qbase + (size_t)(trow + ln) * HDIM;
#pragma unroll
    for (int kk = 0; kk < 2; ++kk) {
      union { v16h v; v8h p[2]; } a;
      a.p[0] = *(const v8h*)(qp + kk * 32 + 16 * hi);
      a.p[1] = *(const v8h*)(qp + kk * 32 + 16 * hi + 8);
      qf[kk] = a.v;
    }
  }

  float mrun[8], lrun[8];
  v8f oacc[4];
  const v8f vz = {};
#pragma unroll
  for (int r = 0; r < 8; ++r) { mrun[r] = NEGBIG; lrun[r] = 0.0f; }
#pragma unroll
  for (int dt = 0; dt < 4; ++dt) oacc[dt] = vz;

  _Float16* pb = Pbuf + w * 16 * 64;
  const int srow = tid >> 2;          // 0..63
  const int sseg = (tid & 3) * 16;    // 0,16,32,48

  for (int s0 = 0; s0 < T_SEQ; s0 += 64) {
    __syncthreads();
    *(v16h*)(Ktile + srow * 64 + sseg) =
        *(const v16h*)(Kbase + (size_t)(s0 + srow) * HDIM + sseg);
    *(v16h*)(Vtile + srow * 64 + sseg) =
        *(const v16h*)(Vbase + (size_t)srow * T_SEQ + s0 + sseg);
    __syncthreads();

    // ---- scores: S = Q . K^T  (16 rows x 64 cols per wave)
    v8f sc[4];
#pragma unroll
    for (int nt = 0; nt < 4; ++nt) sc[nt] = vz;
#pragma unroll
    for (int kk = 0; kk < 2; ++kk)
#pragma unroll
      for (int nt = 0; nt < 4; ++nt) {
        v16h bf = *(const v16h*)(Ktile + (nt * 16 + ln) * 64 + kk * 32 + 16 * hi);
        sc[nt] = wmma16(qf[kk], bf, sc[nt]);
      }

    float madd[4];
#pragma unroll
    for (int nt = 0; nt < 4; ++nt)
      madd[nt] = mrow[s0 + nt * 16 + ln] ? NEGBIG : 0.0f;

    // ---- online softmax (row stats replicated across each 16-lane half)
    float mnew[8], psum[8];
#pragma unroll
    for (int r = 0; r < 8; ++r) {
      float v = NEGBIG;
#pragma unroll
      for (int nt = 0; nt < 4; ++nt) v = fmaxf(v, sc[nt][r] + madd[nt]);
#pragma unroll
      for (int msk = 1; msk < 16; msk <<= 1)
        v = fmaxf(v, __shfl_xor(v, msk, 32));
      mnew[r] = fmaxf(mrun[r], v);
      psum[r] = 0.0f;
    }
#pragma unroll
    for (int nt = 0; nt < 4; ++nt)
#pragma unroll
      for (int r = 0; r < 8; ++r) {
        float p = __expf(sc[nt][r] + madd[nt] - mnew[r]);
        psum[r] += p;
        pb[(r + 8 * hi) * 64 + nt * 16 + ln] = (_Float16)p;
      }
#pragma unroll
    for (int r = 0; r < 8; ++r) {
#pragma unroll
      for (int msk = 1; msk < 16; msk <<= 1)
        psum[r] += __shfl_xor(psum[r], msk, 32);
      const float scale = __expf(mrun[r] - mnew[r]);
      lrun[r] = lrun[r] * scale + psum[r];
      mrun[r] = mnew[r];
#pragma unroll
      for (int dt = 0; dt < 4; ++dt) oacc[dt][r] *= scale;
    }

    // ---- O += P . V   (P from per-wave LDS, V^T tile from LDS)
#pragma unroll
    for (int kk = 0; kk < 2; ++kk) {
      union { v16h v; v8h p[2]; } a;
      const _Float16* ap = pb + ln * 64 + kk * 32;
      a.p[0] = *(const v8h*)(ap + 16 * hi);
      a.p[1] = *(const v8h*)(ap + 16 * hi + 8);
#pragma unroll
      for (int dt = 0; dt < 4; ++dt) {
        v16h vf = *(const v16h*)(Vtile + (dt * 16 + ln) * 64 + kk * 32 + 16 * hi);
        oacc[dt] = wmma16(a.v, vf, oacc[dt]);
      }
    }
  }

  // ---- epilogue: attn[t, b, h*64+d] = O / l
  float linv[8];
#pragma unroll
  for (int r = 0; r < 8; ++r) linv[r] = 1.0f / lrun[r];
#pragma unroll
  for (int dt = 0; dt < 4; ++dt)
#pragma unroll
    for (int r = 0; r < 8; ++r) {
      const int t = trow + r + 8 * hi;
      const int e = h * 64 + dt * 16 + ln;
      Attnh[(size_t)(t * BATCH + b) * EMB + e] = (_Float16)(oacc[dt][r] * linv[r]);
    }
}

// ---------------------------------------------------------------- output projection
__global__ __launch_bounds__(256) void outproj_kernel(
    const _Float16* __restrict__ Ah, const _Float16* __restrict__ Woh,
    const float* __restrict__ b_out, float* __restrict__ out) {
  __shared__ __align__(64) _Float16 lA[128 * 32];
  __shared__ __align__(64) _Float16 lB[128 * 32];
  const int m0 = blockIdx.x * 128;
  const int n0 = blockIdx.y * 128;
  v8f acc[4][2];
  gemm_block(Ah, Woh, EMB, m0, n0, lA, lB, acc);

  const int tid = threadIdx.x, w = tid >> 5, lane = tid & 31;
  const int hi = lane >> 4, ln = lane & 15;
  const int wm = (w >> 2) * 64, wn = (w & 3) * 32;
#pragma unroll
  for (int mt = 0; mt < 4; ++mt)
#pragma unroll
    for (int nt = 0; nt < 2; ++nt)
#pragma unroll
      for (int r = 0; r < 8; ++r) {
        const int m = m0 + wm + mt * 16 + r + 8 * hi;
        const int f = n0 + wn + nt * 16 + ln;
        out[(size_t)m * EMB + f] = acc[mt][nt][r] + b_out[f];
      }
}

// ---------------------------------------------------------------- launcher
extern "C" void kernel_launch(void* const* d_in, const int* in_sizes, int n_in,
                              void* d_out, int out_size, void* d_ws, size_t ws_size,
                              hipStream_t stream) {
  const float*         query = (const float*)d_in[0];
  const unsigned char* mask  = (const unsigned char*)d_in[1]; // bool [B,T]
  const float*         w_in  = (const float*)d_in[2];
  const float*         b_in  = (const float*)d_in[3];
  const float*         w_out = (const float*)d_in[4];
  const float*         b_out = (const float*)d_in[5];
  float* out = (float*)d_out;

  char* ws = (char*)d_ws;
  size_t o = 0;
  _Float16* Xh  = (_Float16*)(ws + o); o += (size_t)MROWS * EMB * 2;       // 16 MiB
  _Float16* Wih = (_Float16*)(ws + o); o += (size_t)3 * EMB * EMB * 2;     //  6 MiB
  _Float16* Woh = (_Float16*)(ws + o); o += (size_t)EMB * EMB * 2;         //  2 MiB
  _Float16* Qh  = (_Float16*)(ws + o); o += (size_t)MROWS * EMB * 2;
  _Float16* Kh  = (_Float16*)(ws + o); o += (size_t)MROWS * EMB * 2;
  _Float16* Vth = (_Float16*)(ws + o); o += (size_t)MROWS * EMB * 2;
  _Float16* Ah  = (_Float16*)(ws + o); o += (size_t)MROWS * EMB * 2;

  const int nX  = MROWS * EMB;      // 8388608
  const int nWi = 3 * EMB * EMB;    // 3145728
  const int nWo = EMB * EMB;        // 1048576
  cvt_f32_to_f16<<<(nX  + 255) / 256, 256, 0, stream>>>(query, Xh,  nX);
  cvt_f32_to_f16<<<(nWi + 255) / 256, 256, 0, stream>>>(w_in,  Wih, nWi);
  cvt_f32_to_f16<<<(nWo + 255) / 256, 256, 0, stream>>>(w_out, Woh, nWo);

  // QKV projection: M=8192, N=3072, K=1024
  proj_kernel<<<dim3(MROWS / 128, (3 * EMB) / 128), 256, 0, stream>>>(
      Xh, Wih, b_in, Qh, Kh, Vth);

  // Fused attention: one WG per (b*h, 128 query rows)
  attn_kernel<<<dim3(BATCH * HEADS, T_SEQ / 128), 256, 0, stream>>>(
      Qh, Kh, Vth, mask, Ah);

  // Output projection: M=8192, N=1024, K=1024
  outproj_kernel<<<dim3(MROWS / 128, EMB / 128), 256, 0, stream>>>(
      Ah, Woh, b_out, out);
}
